// HungarianMatcher_30846455119893
// MI455X (gfx1250) — matrix operand order repaired and checked by reading, not compiled, not run on hardware
//
#include <hip/hip_runtime.h>
#include <hip/hip_bf16.h>

typedef __attribute__((ext_vector_type(16))) _Float16 v16h;
typedef __attribute__((ext_vector_type(8)))  float    v8f;

#define BS   4
#define NQ   100
#define NT   25
#define NCC  81
#define NP   65536
#define QTILES 7          // ceil(100/16)
#define QPAD  112
#define TPAD  32
#define NS    64          // K-slices over P
#define KSTEP 32          // WMMA K per step (f16)
#define KSLICE (NP / NS)          // 1024
#define NSTEPS (KSLICE / KSTEP)   // 32

// workspace layout (float offsets)
#define SZ_XY   (NS * BS * QPAD * TPAD)   // 917504
#define OFF_XY  0
#define OFF_SY  (OFF_XY + SZ_XY)
#define SZ_SP   (NS * BS * QPAD)          // 28672
#define OFF_SP  (OFF_SY + SZ_XY)
#define OFF_SIG (OFF_SP + SZ_SP)
#define SZ_YS   (NS * BS * TPAD)          // 8192
#define OFF_YS  (OFF_SIG + SZ_SP)
#define OFF_CLS (OFF_YS + SZ_YS)          // + 10000

__device__ __forceinline__ float sig_and_softplus(float f, float& sp_acc) {
  // e = exp(-|f|); r = 1/(1+e) = sigmoid(|f|)
  float e = __expf(-fabsf(f));
  float r = __builtin_amdgcn_rcpf(1.0f + e);
  float s = (f >= 0.0f) ? r : (1.0f - r);
  // softplus(f) = max(f,0) + log1p(e) = max(f,0) - log(r)
  sp_acc += fmaxf(f, 0.0f) - __logf(r);
  return s;
}

// convert one float4 of A (x values), track softplus/sigmoid sums, fill f16 vecs
#define DO_A(F4, B0)                                                            \
  { float f, s;                                                                 \
    f = (F4).x * qs; s = sig_and_softplus(f, sp_acc); sg_acc += s;              \
    ax[(B0)+0] = (_Float16)f; as[(B0)+0] = (_Float16)s;                         \
    f = (F4).y * qs; s = sig_and_softplus(f, sp_acc); sg_acc += s;              \
    ax[(B0)+1] = (_Float16)f; as[(B0)+1] = (_Float16)s;                         \
    f = (F4).z * qs; s = sig_and_softplus(f, sp_acc); sg_acc += s;              \
    ax[(B0)+2] = (_Float16)f; as[(B0)+2] = (_Float16)s;                         \
    f = (F4).w * qs; s = sig_and_softplus(f, sp_acc); sg_acc += s;              \
    ax[(B0)+3] = (_Float16)f; as[(B0)+3] = (_Float16)s; }

// convert one float4 of B (y values), track y-sum
#define DO_B(F4, B0, VEC, SC, YS)                                               \
  { float f;                                                                    \
    f = (F4).x * (SC); (VEC)[(B0)+0] = (_Float16)f; (YS) += f;                  \
    f = (F4).y * (SC); (VEC)[(B0)+1] = (_Float16)f; (YS) += f;                  \
    f = (F4).z * (SC); (VEC)[(B0)+2] = (_Float16)f; (YS) += f;                  \
    f = (F4).w * (SC); (VEC)[(B0)+3] = (_Float16)f; (YS) += f; }

__global__ __launch_bounds__(128)
void hm_pair_gemm_kernel(const float* __restrict__ x,   // [BS,NQ,NP] mask logits
                         const float* __restrict__ y,   // [BS,NT,NP] target masks
                         float* __restrict__ ws)
{
  const int lane  = threadIdx.x & 31;
  const int wave  = threadIdx.x >> 5;
  const int qtile = blockIdx.x;                 // 0..6
  const int b     = blockIdx.y;                 // 0..3
  const int ks    = blockIdx.z * 4 + wave;      // 0..63  K-slice

  const int row = lane & 15;                    // M (A/C rows) or N (B/C cols)
  const int hi  = lane >> 4;                    // K-half selector

  const int  q   = qtile * 16 + row;
  const bool qv  = (q < NQ);
  const int  qc  = qv ? q : (NQ - 1);
  const float qs = qv ? 1.0f : 0.0f;

  const int  t1   = 16 + row;
  const bool t1v  = (t1 < NT);
  const int  t1c  = t1v ? t1 : (NT - 1);
  const float ts1 = t1v ? 1.0f : 0.0f;

  const float* xrow  = x + (size_t)(b * NQ + qc)  * NP;
  const float* yrow0 = y + (size_t)(b * NT + row) * NP;   // t-tile 0: t=row<16<NT
  const float* yrow1 = y + (size_t)(b * NT + t1c) * NP;   // t-tile 1

  v8f acc_xy0 = {}, acc_xy1 = {}, acc_sy0 = {}, acc_sy1 = {};
  float sp_acc = 0.0f, sg_acc = 0.0f, ys0_acc = 0.0f, ys1_acc = 0.0f;

  int p0 = ks * KSLICE;
  for (int s = 0; s < NSTEPS; ++s, p0 += KSTEP) {
    // ---- A operand: 16 f16/lane.  elems 0..15 <-> K = hi*16+0..15
    const float* xp = xrow + p0 + hi * 16;
    float4 a0 = *(const float4*)(xp + 0);
    float4 a1 = *(const float4*)(xp + 4);
    float4 a2 = *(const float4*)(xp + 8);
    float4 a3 = *(const float4*)(xp + 12);
    v16h ax = {}, as = {};
    DO_A(a0, 0) DO_A(a1, 4) DO_A(a2, 8) DO_A(a3, 12)

    // ---- B operands: lanes 0-15 (N=lane): K=0..15; lanes 16-31: K=16..31
    const float* yp0 = yrow0 + p0 + hi * 16;
    float4 c0 = ((const float4*)yp0)[0];
    float4 c1 = ((const float4*)yp0)[1];
    float4 c2 = ((const float4*)yp0)[2];
    float4 c3 = ((const float4*)yp0)[3];
    v16h vb0 = {};
    DO_B(c0, 0, vb0, 1.0f, ys0_acc) DO_B(c1, 4,  vb0, 1.0f, ys0_acc)
    DO_B(c2, 8, vb0, 1.0f, ys0_acc) DO_B(c3, 12, vb0, 1.0f, ys0_acc)

    const float* yp1 = yrow1 + p0 + hi * 16;
    float4 d0 = ((const float4*)yp1)[0];
    float4 d1 = ((const float4*)yp1)[1];
    float4 d2 = ((const float4*)yp1)[2];
    float4 d3 = ((const float4*)yp1)[3];
    v16h vb1 = {};
    DO_B(d0, 0, vb1, ts1, ys1_acc) DO_B(d1, 4,  vb1, ts1, ys1_acc)
    DO_B(d2, 8, vb1, ts1, ys1_acc) DO_B(d3, 12, vb1, ts1, ys1_acc)

    // ---- 4 WMMAs: D = A*B + C,  f32 accumulators
    acc_xy0 = __builtin_amdgcn_wmma_f32_16x16x32_f16(false, ax, false, vb0,
                                                     (short)0, acc_xy0, false, false);
    acc_xy1 = __builtin_amdgcn_wmma_f32_16x16x32_f16(false, ax, false, vb1,
                                                     (short)0, acc_xy1, false, false);
    acc_sy0 = __builtin_amdgcn_wmma_f32_16x16x32_f16(false, as, false, vb0,
                                                     (short)0, acc_sy0, false, false);
    acc_sy1 = __builtin_amdgcn_wmma_f32_16x16x32_f16(false, as, false, vb1,
                                                     (short)0, acc_sy1, false, false);
  }

  // ---- write pairwise partials.  C/D layout: VGPR r, lane: M=r+8*hi, N=row
  float* pxy = ws + OFF_XY;
  float* psy = ws + OFF_SY;
#pragma unroll
  for (int r = 0; r < 8; ++r) {
    const int M = r + 8 * hi;
    const int idx = ((ks * BS + b) * QPAD + qtile * 16 + M) * TPAD + row;
    pxy[idx]      = acc_xy0[r];
    pxy[idx + 16] = acc_xy1[r];
    psy[idx]      = acc_sy0[r];
    psy[idx + 16] = acc_sy1[r];
  }

  // ---- per-query reductions: combine the two K-halves held by lane and lane^16
  sp_acc += __shfl_xor(sp_acc, 16, 32);
  sg_acc += __shfl_xor(sg_acc, 16, 32);
  if (hi == 0) {
    ws[OFF_SP  + (ks * BS + b) * QPAD + qtile * 16 + row] = sp_acc;
    ws[OFF_SIG + (ks * BS + b) * QPAD + qtile * 16 + row] = sg_acc;
  }

  // ---- per-target y sums (only q-tile 0 contributes; slices partition P)
  ys0_acc += __shfl_xor(ys0_acc, 16, 32);
  ys1_acc += __shfl_xor(ys1_acc, 16, 32);
  if (qtile == 0 && hi == 0) {
    ws[OFF_YS + (ks * BS + b) * TPAD + row]      = ys0_acc;
    ws[OFF_YS + (ks * BS + b) * TPAD + 16 + row] = ys1_acc;
  }
}

__global__ void hm_class_kernel(const float* __restrict__ logits,  // [BS,NQ,NCC]
                                const int*   __restrict__ labels,  // [BS,NT]
                                float* __restrict__ ws)
{
  int i = blockIdx.x * blockDim.x + threadIdx.x;
  if (i >= BS * NQ * NT) return;
  const int t = i % NT;
  const int q = (i / NT) % NQ;
  const int b = i / (NT * NQ);
  const float* L = logits + (b * NQ + q) * NCC;
  float mx = -3.0e38f;
  for (int c = 0; c < NCC; ++c) mx = fmaxf(mx, L[c]);
  float sum = 0.0f;
  for (int c = 0; c < NCC; ++c) sum += __expf(L[c] - mx);
  int lab = labels[b * NT + t];
  lab = lab < 0 ? 0 : (lab >= NCC ? NCC - 1 : lab);
  float prob = __expf(L[lab] - mx) / sum;
  ws[OFF_CLS + i] = -prob;   // COST_CLASS = 1
}

__global__ void hm_combine_kernel(const float* __restrict__ ws,
                                  float* __restrict__ out)
{
  int i = blockIdx.x * blockDim.x + threadIdx.x;
  if (i >= BS * NQ * NT) return;
  const int t = i % NT;
  const int q = (i / NT) % NQ;
  const int b = i / (NT * NQ);

  float xy = 0.0f, sy = 0.0f, sp = 0.0f, sg = 0.0f, ysum = 0.0f;
  for (int ks = 0; ks < NS; ++ks) {           // fixed order -> deterministic
    const int base = (ks * BS + b);
    const int idx  = (base * QPAD + q) * TPAD + t;
    xy   += ws[OFF_XY  + idx];
    sy   += ws[OFF_SY  + idx];
    sp   += ws[OFF_SP  + base * QPAD + q];
    sg   += ws[OFF_SIG + base * QPAD + q];
    ysum += ws[OFF_YS  + base * TPAD + t];
  }
  const float invP = 1.0f / (float)NP;
  const float bce  = sp * invP - xy * invP;
  const float dice = 1.0f - 2.0f * sy / (sg + ysum + 1e-6f);
  out[i] = ws[OFF_CLS + i] + bce + dice;      // COST_* = 1
}

extern "C" void kernel_launch(void* const* d_in, const int* in_sizes, int n_in,
                              void* d_out, int out_size, void* d_ws, size_t ws_size,
                              hipStream_t stream) {
  const float* pred_logits = (const float*)d_in[0];   // [4,100,81]
  const float* pred_masks  = (const float*)d_in[1];   // [4,100,256,256]
  const int*   tgt_labels  = (const int*)  d_in[2];   // [4,25]
  const float* tgt_masks   = (const float*)d_in[3];   // [4,25,256,256]
  float* out = (float*)d_out;                         // [4,100,25]
  float* ws  = (float*)d_ws;

  dim3 grid(QTILES, BS, NS / 4);
  hm_pair_gemm_kernel<<<grid, 128, 0, stream>>>(pred_masks, tgt_masks, ws);

  const int n = BS * NQ * NT;
  hm_class_kernel<<<(n + 255) / 256, 256, 0, stream>>>(pred_logits, tgt_labels, ws);
  hm_combine_kernel<<<(n + 255) / 256, 256, 0, stream>>>(ws, out);
}
